// TextDecoderLayer_29738353557823
// MI455X (gfx1250) — matrix operand order repaired and checked by reading, not compiled, not run on hardware
//
#include <hip/hip_runtime.h>

typedef unsigned short u16;
typedef __attribute__((ext_vector_type(16))) __bf16 bf16x16;
typedef __attribute__((ext_vector_type(8)))  float  floatx8;
typedef int gcc_int4 __attribute__((vector_size(16)));   // matches builtin's V4i pointee

union BF16Frag { bf16x16 v; uint4 q[2]; u16 h[16]; };

#define AS1 __attribute__((address_space(1)))
#define AS3 __attribute__((address_space(3)))

#if defined(__has_builtin)
#if __has_builtin(__builtin_amdgcn_global_load_async_to_lds_b128)
#define HAVE_ASYNC 1
#endif
#endif
#ifndef HAVE_ASYNC
#define HAVE_ASYNC 0
#warning "gfx1250 async load-to-LDS builtin not found; falling back to sync LDS copies"
#endif

__device__ __forceinline__ u16 f2bf(float f) {
    unsigned int u = __float_as_uint(f);
    unsigned int r = u + 0x7fffu + ((u >> 16) & 1u);
    return (u16)(r >> 16);
}

__device__ __forceinline__ floatx8 bwmma(const BF16Frag& a, const BF16Frag& b, floatx8 c) {
    return __builtin_amdgcn_wmma_f32_16x16x32_bf16(false, a.v, false, b.v, (short)0, c, false, false);
}

// 16-byte global -> LDS copy (async on gfx1250 when available)
__device__ __forceinline__ void cp16_async(void* ldsp, const void* gp) {
#if HAVE_ASYNC
    __builtin_amdgcn_global_load_async_to_lds_b128(
        (AS1 gcc_int4*)gp, (AS3 gcc_int4*)ldsp, 0, 0);
#else
    *(uint4*)ldsp = *(const uint4*)gp;
#endif
}

// ---------------------------------------------------------------------------
// Weight convert + transpose: w[K,N] fp32 -> wt[N,K] bf16
// ---------------------------------------------------------------------------
__global__ __launch_bounds__(256) void cvt_transpose_kernel(
    const float* __restrict__ w, u16* __restrict__ wt, int Kd, int Nd)
{
    int i = blockIdx.x * 256 + threadIdx.x;
    int total = Kd * Nd;
    if (i >= total) return;
    int k = i % Kd;
    int n = i / Kd;
    wt[i] = f2bf(w[(size_t)k * Nd + n]);
}

// ---------------------------------------------------------------------------
// Row RMSNorm (row length Dd) -> bf16
// ---------------------------------------------------------------------------
__global__ __launch_bounds__(256) void rmsnorm_kernel(
    const float* __restrict__ x, const float* __restrict__ scale,
    u16* __restrict__ out, int Dd)
{
    __shared__ float red[8];
    const int row = blockIdx.x;
    const float* xr = x + (size_t)row * Dd;
    float ss = 0.f;
    for (int i = threadIdx.x; i < Dd; i += 256) { float v = xr[i]; ss += v * v; }
    for (int off = 16; off > 0; off >>= 1) ss += __shfl_xor(ss, off, 32);
    if ((threadIdx.x & 31) == 0) red[threadIdx.x >> 5] = ss;
    __syncthreads();
    float tot = red[0] + red[1] + red[2] + red[3] + red[4] + red[5] + red[6] + red[7];
    float rr = rsqrtf(tot / (float)Dd + 1e-6f);
    u16* orow = out + (size_t)row * Dd;
    for (int i = threadIdx.x; i < Dd; i += 256)
        orow[i] = f2bf(xr[i] * rr * scale[i]);
}

// ---------------------------------------------------------------------------
// Per-head RMSNorm + RoPE.  One wave per (b,t,head).  Writes [b,head,t,k] bf16.
// ---------------------------------------------------------------------------
__global__ __launch_bounds__(32) void ropenorm_kernel(
    const float* __restrict__ proj,     // [B*T, NH*64]
    const float* __restrict__ nscale,   // [64]
    const float* __restrict__ sinp,     // [B*T, 32]
    const float* __restrict__ cosp,     // [B*T, 32]
    u16* __restrict__ out,              // [(b*NH+h)*T + t]*64 + k
    int NH, int T)
{
    const int h  = blockIdx.x % NH;
    const int bt = blockIdx.x / NH;
    const int b  = bt / T;
    const int t  = bt % T;
    const int i  = threadIdx.x;              // 0..31
    const float* src = proj + (size_t)bt * NH * 64 + h * 64;
    float x1 = src[i], x2 = src[i + 32];
    float ss = x1 * x1 + x2 * x2;
    for (int off = 16; off > 0; off >>= 1) ss += __shfl_xor(ss, off, 32);
    float rr = rsqrtf(ss / 64.f + 1e-6f);
    x1 *= rr * nscale[i];
    x2 *= rr * nscale[i + 32];
    float s = sinp[(size_t)bt * 32 + i];
    float c = cosp[(size_t)bt * 32 + i];
    size_t base = (((size_t)b * NH + h) * T + t) * 64;
    out[base + i]      = f2bf(x1 * c - x2 * s);
    out[base + 32 + i] = f2bf(x2 * c + x1 * s);
}

// ---------------------------------------------------------------------------
// V transpose: vf [B*T, G*64] fp32 -> vtb [(b*G+g)*64 + k]*T + t  bf16
// ---------------------------------------------------------------------------
__global__ __launch_bounds__(256) void vtrans_kernel(
    const float* __restrict__ vf, u16* __restrict__ vtb, int T, int G)
{
    int i = blockIdx.x * 256 + threadIdx.x;        // flat over output
    int t  = i % T;
    int k  = (i / T) % 64;
    int bg = i / (T * 64);
    int b  = bg / G;
    int g  = bg % G;
    vtb[i] = f2bf(vf[((size_t)b * T + t) * (G * 64) + g * 64 + k]);
}

// ---------------------------------------------------------------------------
// bf16 GEMM, f32 accumulate via v_wmma_f32_16x16x32_bf16.
// Block: 256 threads (8 waves), tile BM=64 x BN=128, BK=32.
// Double-buffered LDS staging via async load-to-LDS (ASYNCcnt) + barriers.
// A: [M,K] bf16 row-major.  Bt: [N,K] bf16 (pre-transposed weight).
// Requires M%64==0, N%128==0, K%32==0 (true for all launches here).
// EPI 0: Cf = acc            (fp32)
// EPI 1: Cf = Res + acc      (fp32, fused residual)
// EPI 2: Cb = bf16(silu(acc0) * acc1)   (dual-B gate/up fusion)
// ---------------------------------------------------------------------------
#define LDS_STRIDE 40   // 32 elems + 8 pad; 80B row stride, 16B aligned

template <int EPI>
__global__ __launch_bounds__(256) void gemm_kernel(
    const u16* __restrict__ A, const u16* __restrict__ Bt0, const u16* __restrict__ Bt1,
    float* __restrict__ Cf, u16* __restrict__ Cb, const float* __restrict__ Res,
    int M, int N, int K)
{
    extern __shared__ __align__(16) unsigned char smem[];
    u16* As  = (u16*)smem;                       // [2][64][LDS_STRIDE]
    u16* Bs0 = As + 2 * 64 * LDS_STRIDE;         // [2][128][LDS_STRIDE]
    u16* Bs1 = Bs0 + 2 * 128 * LDS_STRIDE;       // [2][128][LDS_STRIDE] (EPI==2)

    const int lane  = threadIdx.x & 31;
    const int wave  = threadIdx.x >> 5;
    const int waveM = wave >> 1;                 // 0..3
    const int waveN = wave & 1;                  // 0..1
    const int rowBlock = blockIdx.y * 64;
    const int colBlock = blockIdx.x * 128;

    const int r      = lane & 15;
    const int khalfA = (lane >> 4) * 8;          // A layout K half
    const int koffB  = (lane >> 4) * 8;          // B layout K half

    constexpr int NL = (EPI == 2) ? 5 : 3;       // async copies per thread/stage

    floatx8 acc0[4] = {}, acc1[4] = {};

    auto load_stage = [&](int buf, int k0) {
        {   // A tile 64x32 : 1 chunk per thread
            int c = threadIdx.x;
            int row = c >> 2, col = (c & 3) * 8;
            cp16_async(As + ((size_t)buf * 64 + row) * LDS_STRIDE + col,
                       A + (size_t)(rowBlock + row) * K + k0 + col);
        }
#pragma unroll
        for (int i = 0; i < 2; ++i) {            // B tile(s) 128x32 : 2 chunks per thread
            int c = threadIdx.x + i * 256;
            int row = c >> 2, col = (c & 3) * 8;
            cp16_async(Bs0 + ((size_t)buf * 128 + row) * LDS_STRIDE + col,
                       Bt0 + (size_t)(colBlock + row) * K + k0 + col);
            if constexpr (EPI == 2)
                cp16_async(Bs1 + ((size_t)buf * 128 + row) * LDS_STRIDE + col,
                           Bt1 + (size_t)(colBlock + row) * K + k0 + col);
        }
    };

    load_stage(0, 0);
    int buf = 0;
    for (int k0 = 0; k0 < K; k0 += 32) {
        const bool more = (k0 + 32) < K;
        if (more) load_stage(buf ^ 1, k0 + 32);
#if HAVE_ASYNC
        if (more) asm volatile("s_wait_asynccnt %0" :: "n"(NL) : "memory");
        else      asm volatile("s_wait_asynccnt 0" ::: "memory");
#endif
        __syncthreads();                          // stage `buf` visible to all waves

        BF16Frag af;
        const u16* ap = As + ((size_t)buf * 64 + waveM * 16 + r) * LDS_STRIDE + khalfA;
        af.q[0] = *(const uint4*)(ap);
        af.q[1] = *(const uint4*)(ap + 16);
#pragma unroll
        for (int t = 0; t < 4; ++t) {
            const u16* bp = Bs0 + ((size_t)buf * 128 + waveN * 64 + t * 16 + r) * LDS_STRIDE + koffB;
            BF16Frag bf;
            bf.q[0] = *(const uint4*)(bp);
            bf.q[1] = *(const uint4*)(bp + 16);
            acc0[t] = bwmma(af, bf, acc0[t]);
            if constexpr (EPI == 2) {
                const u16* bp1 = Bs1 + ((size_t)buf * 128 + waveN * 64 + t * 16 + r) * LDS_STRIDE + koffB;
                BF16Frag bg;
                bg.q[0] = *(const uint4*)(bp1);
                bg.q[1] = *(const uint4*)(bp1 + 16);
                acc1[t] = bwmma(af, bg, acc1[t]);
            }
        }
        __syncthreads();                          // all waves done with `buf` before refill
        buf ^= 1;
    }

    const int half = lane >> 4;
    const int rowBase = rowBlock + waveM * 16;
    const int colBase = colBlock + waveN * 64;
#pragma unroll
    for (int t = 0; t < 4; ++t) {
#pragma unroll
        for (int v = 0; v < 8; ++v) {
            int row = rowBase + v + 8 * half;
            int col = colBase + t * 16 + r;
            size_t idx = (size_t)row * N + col;
            if constexpr (EPI == 0) {
                Cf[idx] = acc0[t][v];
            } else if constexpr (EPI == 1) {
                Cf[idx] = Res[idx] + acc0[t][v];
            } else {
                float g = acc0[t][v], u = acc1[t][v];
                float s = g / (1.0f + __expf(-g));
                Cb[idx] = f2bf(s * u);
            }
        }
    }
}

// ---------------------------------------------------------------------------
// Flash attention (causal, GQA).  One 16-query tile per wave, 32-key steps.
// qb:  [(b*H+h)*T + t]*64 + k      bf16
// kb:  [(b*G+g)*T + s]*64 + k      bf16
// vtb: [(b*G+g)*64 + k]*T + s      bf16 (V transposed)
// ab:  [b*T + t, H*64]             bf16 attention output
// ---------------------------------------------------------------------------
__global__ __launch_bounds__(256) void flash_kernel(
    const u16* __restrict__ qb, const u16* __restrict__ kb,
    const u16* __restrict__ vtb, u16* __restrict__ ab,
    int T, int H, int G)
{
    __shared__ __align__(16) u16 pshare[8][16 * 32];
    const int lane = threadIdx.x & 31;
    const int wave = threadIdx.x >> 5;
    const int bh = blockIdx.y;
    const int b = bh / H, h = bh % H;
    const int g = h / (H / G);
    const int qBase = blockIdx.x * 128 + wave * 16;
    const int r = lane & 15;
    const int half = lane >> 4;

    const u16* qhead = qb  + (size_t)bh * T * 64;
    const u16* khead = kb  + (size_t)(b * G + g) * T * 64;
    const u16* vhead = vtb + (size_t)(b * G + g) * 64 * T;

    // Q fragments (K-dim 0..31 and 32..63)
    BF16Frag qf[2];
    const u16* qrow = qhead + (size_t)(qBase + r) * 64 + half * 8;
    qf[0].q[0] = *(const uint4*)(qrow);
    qf[0].q[1] = *(const uint4*)(qrow + 16);
    qf[1].q[0] = *(const uint4*)(qrow + 32);
    qf[1].q[1] = *(const uint4*)(qrow + 48);

    float rowmax[8], rowsum[8], alpha[8];
#pragma unroll
    for (int v = 0; v < 8; ++v) { rowmax[v] = -3.0e38f; rowsum[v] = 0.f; }
    floatx8 o[4] = {};

    const int sEnd = qBase + 16;
    for (int sBase = 0; sBase < sEnd; sBase += 32) {
        // scores: two 16x16 tiles covering keys sBase..sBase+31
        floatx8 sc[2] = {};
#pragma unroll
        for (int nt = 0; nt < 2; ++nt) {
            const u16* kr = khead + (size_t)(sBase + nt * 16 + r) * 64 + half * 8;
            BF16Frag kf0, kf1;
            kf0.q[0] = *(const uint4*)(kr);
            kf0.q[1] = *(const uint4*)(kr + 16);
            kf1.q[0] = *(const uint4*)(kr + 32);
            kf1.q[1] = *(const uint4*)(kr + 48);
            sc[nt] = bwmma(qf[0], kf0, sc[nt]);
            sc[nt] = bwmma(qf[1], kf1, sc[nt]);
        }
        // scale, causal mask, online softmax
#pragma unroll
        for (int v = 0; v < 8; ++v) {
            const int m = qBase + v + 8 * half;
#pragma unroll
            for (int nt = 0; nt < 2; ++nt) {
                int s = sBase + nt * 16 + r;
                float val = sc[nt][v] * 0.125f;
                sc[nt][v] = (s <= m) ? val : -3.0e38f;
            }
            float t0 = sc[0][v], t1 = sc[1][v];
            float mloc = fmaxf(t0, t1);
            for (int off = 8; off > 0; off >>= 1)
                mloc = fmaxf(mloc, __shfl_xor(mloc, off, 32));
            float mnew = fmaxf(rowmax[v], mloc);
            float a  = __expf(rowmax[v] - mnew);
            float p0 = __expf(t0 - mnew);
            float p1 = __expf(t1 - mnew);
            float ps = p0 + p1;
            for (int off = 8; off > 0; off >>= 1) ps += __shfl_xor(ps, off, 32);
            rowsum[v] = rowsum[v] * a + ps;
            rowmax[v] = mnew;
            alpha[v]  = a;
            sc[0][v] = p0; sc[1][v] = p1;
        }
        // C-layout -> A-layout via per-wave LDS slab
        u16* pw = pshare[wave];
#pragma unroll
        for (int v = 0; v < 8; ++v) {
            int m = v + 8 * half;
            pw[m * 32 + r]      = f2bf(sc[0][v]);
            pw[m * 32 + 16 + r] = f2bf(sc[1][v]);
        }
        asm volatile("s_wait_dscnt 0" ::: "memory");
        BF16Frag pa;
        const u16* pr = pw + r * 32 + half * 8;
        pa.q[0] = *(const uint4*)(pr);
        pa.q[1] = *(const uint4*)(pr + 16);
        // rescale O and accumulate P @ V
#pragma unroll
        for (int t4 = 0; t4 < 4; ++t4)
#pragma unroll
            for (int v = 0; v < 8; ++v) o[t4][v] *= alpha[v];
#pragma unroll
        for (int t4 = 0; t4 < 4; ++t4) {
            const u16* vr = vhead + (size_t)(t4 * 16 + r) * T + sBase + half * 8;
            BF16Frag vf2;
            vf2.q[0] = *(const uint4*)(vr);
            vf2.q[1] = *(const uint4*)(vr + 16);
            o[t4] = bwmma(pa, vf2, o[t4]);
        }
    }
    // normalize + write bf16 attn output [b*T + m, h*64 + col]
#pragma unroll
    for (int t4 = 0; t4 < 4; ++t4) {
#pragma unroll
        for (int v = 0; v < 8; ++v) {
            int m = qBase + v + 8 * half;
            int col = t4 * 16 + r;
            float val = o[t4][v] / rowsum[v];
            ab[((size_t)b * T + m) * (H * 64) + h * 64 + col] = f2bf(val);
        }
    }
}

// ---------------------------------------------------------------------------
// Host launcher
// ---------------------------------------------------------------------------
extern "C" void kernel_launch(void* const* d_in, const int* in_sizes, int n_in,
                              void* d_out, int out_size, void* d_ws, size_t ws_size,
                              hipStream_t stream)
{
    constexpr int Bc = 2, Tc = 2048, Dc = 1024, Hc = 16, Gc = 4, Fc = 4096;
    constexpr int BT = Bc * Tc;
    constexpr int HK = Hc * 64;   // 1024
    constexpr int GK = Gc * 64;   // 256

    const float* hidden = (const float*)d_in[0];
    const float* sinp   = (const float*)d_in[1];
    const float* cosp   = (const float*)d_in[2];
    const float* ln1    = (const float*)d_in[4];
    const float* ln2    = (const float*)d_in[5];
    const float* qns    = (const float*)d_in[6];
    const float* kns    = (const float*)d_in[7];
    const float* qw     = (const float*)d_in[8];
    const float* kw     = (const float*)d_in[9];
    const float* vw     = (const float*)d_in[10];
    const float* ow     = (const float*)d_in[11];
    const float* gw     = (const float*)d_in[12];
    const float* uw     = (const float*)d_in[13];
    const float* dw     = (const float*)d_in[14];

    char*  base = (char*)d_ws;
    size_t off  = 0;
    auto carve = [&](size_t bytes) -> void* {
        void* p = base + off;
        off = (off + bytes + 255) & ~(size_t)255;
        return p;
    };
    u16* wq_t = (u16*)carve((size_t)Dc * HK * 2);
    u16* wk_t = (u16*)carve((size_t)Dc * GK * 2);
    u16* wv_t = (u16*)carve((size_t)Dc * GK * 2);
    u16* wo_t = (u16*)carve((size_t)HK * Dc * 2);
    u16* wg_t = (u16*)carve((size_t)Dc * Fc * 2);
    u16* wu_t = (u16*)carve((size_t)Dc * Fc * 2);
    u16* wd_t = (u16*)carve((size_t)Fc * Dc * 2);
    u16* xb   = (u16*)carve((size_t)BT * Dc * 2);
    float* qf = (float*)carve((size_t)BT * HK * 4);
    float* kf = (float*)carve((size_t)BT * GK * 4);
    float* vf = (float*)carve((size_t)BT * GK * 4);
    u16* qbuf = (u16*)carve((size_t)BT * HK * 2);
    u16* kbuf = (u16*)carve((size_t)BT * GK * 2);
    u16* vtb  = (u16*)carve((size_t)BT * GK * 2);
    u16* abuf = (u16*)carve((size_t)BT * HK * 2);
    float* h2 = (float*)carve((size_t)BT * Dc * 4);
    u16* yb   = (u16*)carve((size_t)BT * Dc * 2);
    u16* hb   = (u16*)carve((size_t)BT * Fc * 2);
    float* outp = (float*)d_out;

    auto cvtGrid = [](int n) { return dim3((n + 255) / 256); };
    constexpr size_t smemSingle = (size_t)(2 * 64 * LDS_STRIDE + 2 * 128 * LDS_STRIDE) * 2;
    constexpr size_t smemDual   = (size_t)(2 * 64 * LDS_STRIDE + 2 * 2 * 128 * LDS_STRIDE) * 2;

    // 1) weight convert + transpose -> bf16 [N,K]
    cvt_transpose_kernel<<<cvtGrid(Dc * HK), 256, 0, stream>>>(qw, wq_t, Dc, HK);
    cvt_transpose_kernel<<<cvtGrid(Dc * GK), 256, 0, stream>>>(kw, wk_t, Dc, GK);
    cvt_transpose_kernel<<<cvtGrid(Dc * GK), 256, 0, stream>>>(vw, wv_t, Dc, GK);
    cvt_transpose_kernel<<<cvtGrid(HK * Dc), 256, 0, stream>>>(ow, wo_t, HK, Dc);
    cvt_transpose_kernel<<<cvtGrid(Dc * Fc), 256, 0, stream>>>(gw, wg_t, Dc, Fc);
    cvt_transpose_kernel<<<cvtGrid(Dc * Fc), 256, 0, stream>>>(uw, wu_t, Dc, Fc);
    cvt_transpose_kernel<<<cvtGrid(Fc * Dc), 256, 0, stream>>>(dw, wd_t, Fc, Dc);

    // 2) x = rmsnorm(hidden, ln1)  -> bf16
    rmsnorm_kernel<<<dim3(BT), 256, 0, stream>>>(hidden, ln1, xb, Dc);

    // 3) q/k/v projections (fp32 out)
    gemm_kernel<0><<<dim3(HK / 128, BT / 64), 256, smemSingle, stream>>>(
        xb, wq_t, nullptr, qf, nullptr, nullptr, BT, HK, Dc);
    gemm_kernel<0><<<dim3(GK / 128, BT / 64), 256, smemSingle, stream>>>(
        xb, wk_t, nullptr, kf, nullptr, nullptr, BT, GK, Dc);
    gemm_kernel<0><<<dim3(GK / 128, BT / 64), 256, smemSingle, stream>>>(
        xb, wv_t, nullptr, vf, nullptr, nullptr, BT, GK, Dc);

    // 4) head rmsnorm + rope -> bf16 layouts; transpose V
    ropenorm_kernel<<<dim3(BT * Hc), 32, 0, stream>>>(qf, qns, sinp, cosp, qbuf, Hc, Tc);
    ropenorm_kernel<<<dim3(BT * Gc), 32, 0, stream>>>(kf, kns, sinp, cosp, kbuf, Gc, Tc);
    vtrans_kernel<<<cvtGrid(BT * GK), 256, 0, stream>>>(vf, vtb, Tc, Gc);

    // 5) flash attention -> bf16 attn [BT, HK]
    flash_kernel<<<dim3(Tc / 128, Bc * Hc), 256, 0, stream>>>(qbuf, kbuf, vtb, abuf, Tc, Hc, Gc);

    // 6) o-projection + residual -> h2 (fp32)
    gemm_kernel<1><<<dim3(Dc / 128, BT / 64), 256, smemSingle, stream>>>(
        abuf, wo_t, nullptr, h2, nullptr, hidden, BT, Dc, HK);

    // 7) y = rmsnorm(h2, ln2) -> bf16
    rmsnorm_kernel<<<dim3(BT), 256, 0, stream>>>(h2, ln2, yb, Dc);

    // 8) fused gate/up GEMM + silu*up -> bf16 hb [BT, F]
    gemm_kernel<2><<<dim3(Fc / 128, BT / 64), 256, smemDual, stream>>>(
        yb, wg_t, wu_t, nullptr, hb, nullptr, BT, Fc, Dc);

    // 9) down projection + residual -> d_out (fp32)
    gemm_kernel<1><<<dim3(Dc / 128, BT / 64), 256, smemSingle, stream>>>(
        hb, wd_t, nullptr, outp, nullptr, h2, BT, Dc, Fc);

    (void)in_sizes; (void)n_in; (void)out_size; (void)ws_size;
}
